// DeformGCN_30305289241172
// MI455X (gfx1250) — matrix operand order repaired and checked by reading, not compiled, not run on hardware
//
#include <hip/hip_runtime.h>
#include <math.h>

// ---------------- problem constants (from reference) ----------------
#define BB 16
#define NN 2048
#define CIN 1475
#define EE 12288
#define MROWS (BB * NN)          // 32768
#define DFEAT (NN * 3)           // 6144
#define NEG_SLOPE 0.01f

typedef __bf16 bf16;
typedef __attribute__((ext_vector_type(8)))  bf16  v8bf;
typedef __attribute__((ext_vector_type(16))) bf16  v16bf;
typedef __attribute__((ext_vector_type(8)))  float v8f;

union Frag { v16bf v; v8bf h[2]; };

// LDS B-tile row stride (elements): 64 K + 8 pad -> 144B rows = 36 banks,
// conflict-free across 16 rows, 16B-aligned chunks.
#define BSTRIDE 72

// ---------------- helpers ----------------
__device__ __forceinline__ unsigned short f2bf_rne(float f) {
    unsigned int u = __float_as_uint(f);
    unsigned int r = u + 0x7FFFu + ((u >> 16) & 1u);
    return (unsigned short)(r >> 16);
}

// ---------------- degree kernels ----------------
__global__ void k_deg_zero(float* deg) {
    int i = blockIdx.x * blockDim.x + threadIdx.x;
    if (i < NN) deg[i] = 0.0f;
}
__global__ void k_deg_count(const int* __restrict__ edges, float* deg) {
    int e = blockIdx.x * blockDim.x + threadIdx.x;
    if (e < EE) atomicAdd(&deg[edges[EE + e]], 1.0f);
}
__global__ void k_deg_fin(const float* __restrict__ deg, float* isq, float* idg) {
    int i = blockIdx.x * blockDim.x + threadIdx.x;
    if (i < NN) {
        float d = deg[i] + 1.0f;          // self loop
        isq[i] = rsqrtf(d);
        idg[i] = 1.0f / d;
    }
}

// ---------------- f32 -> bf16 activation convert (optional LeakyReLU), zero pad K ----------------
__global__ void k_convert_act(const float* __restrict__ src, unsigned short* __restrict__ dst,
                              long long M, int Cin, int Kp, int leaky) {
    long long total = M * (long long)Kp;
    long long stride = (long long)gridDim.x * blockDim.x;
    for (long long i = blockIdx.x * (long long)blockDim.x + threadIdx.x; i < total; i += stride) {
        long long m = i / Kp;
        int k = (int)(i - m * Kp);
        float v = 0.0f;
        if (k < Cin) {
            v = src[m * Cin + k];
            if (leaky && v < 0.0f) v *= NEG_SLOPE;
        }
        dst[i] = f2bf_rne(v);
    }
}

// ---------------- W[cin][cout] f32 -> BT[Npad][Kp] bf16 (transposed, zero padded) ----------------
__global__ void k_convert_wT(const float* __restrict__ W, unsigned short* __restrict__ BT,
                             int cin, int cout, int Kp, int Npad) {
    long long total = (long long)Npad * Kp;
    long long stride = (long long)gridDim.x * blockDim.x;
    for (long long i = blockIdx.x * (long long)blockDim.x + threadIdx.x; i < total; i += stride) {
        int n = (int)(i / Kp);
        int k = (int)(i - (long long)n * Kp);
        float v = (n < cout && k < cin) ? W[(size_t)k * cout + n] : 0.0f;
        BT[i] = f2bf_rne(v);
    }
}

// ---------------- bf16 WMMA GEMM, async-LDS double-buffered B staging ----------------
// A:  [M x Kp] bf16 row-major,  BT: [Npad x Kp] bf16 row-major (B transposed),
// Kp % 64 == 0, Npad = roundup(Np,128), BT rows [Np,Npad) are zero.
// Hw: [M x Np] f32 = A@B.   mode==1: also agg = Hw*inv_deg[row%NN] + bias[col]
// block = 256 threads (8 waves); block tile 128(M) x 128(N); wave w -> rows m0 = bx*128 + w*16.
__global__ __launch_bounds__(256)
void k_gemm_wmma(const unsigned short* __restrict__ A, const unsigned short* __restrict__ BT,
                 float* __restrict__ Hw, float* __restrict__ agg,
                 const float* __restrict__ bias, const float* __restrict__ invdeg,
                 int M, int Np, int Kp, int cout, int mode)
{
    __shared__ unsigned short sB[2][128 * BSTRIDE];   // 2 x 18 KB

    const int tid  = threadIdx.x;
    const int wave = tid >> 5;
    const int lane = tid & 31;
    const int m0 = blockIdx.x * 128 + wave * 16;
    const int n0 = blockIdx.y * 128;
    // scalarize guards so WMMA never runs under a lane mask (EXEC must be all-ones)
    const bool active = (__builtin_amdgcn_readfirstlane(m0) < M);
    const bool haveHi = (n0 + 64) < Np;
    const bool hiHalf = lane >= 16;
    const int  lr = lane & 15;

    const bf16* __restrict__ Ab = (const bf16*)A;
    const bf16* arow = Ab + (size_t)((m0 < M ? m0 : 0) + lr) * Kp;
    const int aoff = hiHalf ? 16 : 0;      // A frag: hi lanes hold K+16
    const int boff = hiHalf ? 16 : 0;      // B frag: hi lanes hold K+16

    // Staging: 128 rows x 64 K bf16 (16 KB) per stage; 1024 chunks of 16B; 4 per thread.
    int rr[4], cc[4];
    const unsigned short* gB[4];
    unsigned ls[4];
#pragma unroll
    for (int c = 0; c < 4; ++c) {
        int j = tid + 256 * c;
        rr[c] = j >> 3; cc[c] = j & 7;
        gB[c] = BT + (size_t)(n0 + rr[c]) * Kp + cc[c] * 8;
        ls[c] = rr[c] * BSTRIDE + cc[c] * 8;
    }

    const int nStages = Kp >> 6;

    // issue stage 0
#pragma unroll
    for (int c = 0; c < 4; ++c) {
        unsigned l = (unsigned)(uintptr_t)&sB[0][ls[c]];
        unsigned long long g = (unsigned long long)(uintptr_t)gB[c];
        asm volatile("global_load_async_to_lds_b128 %0, %1, off" :: "v"(l), "v"(g) : "memory");
    }

    v8f accL[4] = {{}, {}, {}, {}};
    v8f accH[4] = {{}, {}, {}, {}};

    for (int s = 0; s < nStages; ++s) {
        const int cur = s & 1;
        if (s + 1 < nStages) {
            const int nxt = cur ^ 1;
            const size_t go = (size_t)(s + 1) * 64;   // K advance (elements)
#pragma unroll
            for (int c = 0; c < 4; ++c) {
                unsigned l = (unsigned)(uintptr_t)&sB[nxt][ls[c]];
                unsigned long long g = (unsigned long long)(uintptr_t)(gB[c] + go);
                asm volatile("global_load_async_to_lds_b128 %0, %1, off" :: "v"(l), "v"(g) : "memory");
            }
            // 4 just-issued copies stay in flight; current stage retired
            asm volatile("s_wait_asynccnt 0x4" ::: "memory");
        } else {
            asm volatile("s_wait_asynccnt 0x0" ::: "memory");
        }
        __syncthreads();

        if (active) {
            const bf16* ak = arow + (size_t)s * 64;
            const bf16* sb = (const bf16*)&sB[cur][0];
            if (s + 1 < nStages) __builtin_prefetch((const void*)(ak + 64), 0, 3);
#pragma unroll
            for (int kk = 0; kk < 64; kk += 32) {
                Frag af;
                af.h[0] = *(const v8bf*)(ak + kk + aoff);        // K = kk+{0..7}  (+16 hi)
                af.h[1] = *(const v8bf*)(ak + kk + 8 + aoff);    // K = kk+{8..15} (+16 hi)
                const int kb = kk + boff;
#pragma unroll
                for (int t = 0; t < 4; ++t) {
                    Frag bf;
                    bf.h[0] = *(const v8bf*)(sb + (t * 16 + lr) * BSTRIDE + kb);
                    bf.h[1] = *(const v8bf*)(sb + (t * 16 + lr) * BSTRIDE + kb + 8);
                    accL[t] = __builtin_amdgcn_wmma_f32_16x16x32_bf16(false, af.v, false, bf.v,
                                                                     (short)0, accL[t], false, false);
                }
                if (haveHi) {
#pragma unroll
                    for (int t = 0; t < 4; ++t) {
                        Frag bf;
                        bf.h[0] = *(const v8bf*)(sb + ((t + 4) * 16 + lr) * BSTRIDE + kb);
                        bf.h[1] = *(const v8bf*)(sb + ((t + 4) * 16 + lr) * BSTRIDE + kb + 8);
                        accH[t] = __builtin_amdgcn_wmma_f32_16x16x32_bf16(false, af.v, false, bf.v,
                                                                         (short)0, accH[t], false, false);
                    }
                }
            }
        }
        __syncthreads();   // protect buf[cur] from being overwritten by next issue
    }

    if (!active) return;

    // D layout: element r of acc -> row m0 + (hiHalf?8:0) + r, col n0 + t*16 + lr
    const int rbase = m0 + (hiHalf ? 8 : 0);
    const int nTiles = haveHi ? 8 : 4;
#pragma unroll
    for (int t = 0; t < 8; ++t) {
        if (t >= nTiles) break;
        v8f a = (t < 4) ? accL[t] : accH[t - 4];
        int colg = n0 + t * 16 + lr;
        float bb = 0.0f;
        if (mode && colg < cout) bb = bias[colg];
#pragma unroll
        for (int r = 0; r < 8; ++r) {
            int row = rbase + r;
            size_t idx = (size_t)row * Np + colg;
            float v = a[r];
            Hw[idx] = v;
            if (mode) {
                int node = row & (NN - 1);
                agg[idx] = v * invdeg[node] + bb;
            }
        }
    }
}

// ---------------- edge scatter: agg[b,dst,:] += Hw[b,src,:] * norm(e) ----------------
__global__ void k_scatter(const float* __restrict__ Hw, float* __restrict__ agg,
                          const int* __restrict__ edges, const float* __restrict__ isq, int Np)
{
    int e = blockIdx.x;
    int s = edges[e];
    int d = edges[EE + e];
    float w = isq[s] * isq[d];
    int per = BB * Np;
    for (int i = threadIdx.x; i < per; i += blockDim.x) {
        int b = i / Np;
        int f = i - b * Np;
        float v = Hw[((size_t)b * NN + s) * Np + f] * w;
        atomicAdd(&agg[((size_t)b * NN + d) * Np + f], v);
    }
}

// ---------------- final feature pack: feat[b][node*3+c] = leaky(agg5[b,node,c]), bf16 ----------------
__global__ void k_convert_feat(const float* __restrict__ agg5, unsigned short* __restrict__ featbf) {
    int total = BB * DFEAT;
    for (int i = blockIdx.x * blockDim.x + threadIdx.x; i < total; i += gridDim.x * blockDim.x) {
        int b = i / DFEAT;
        int j = i - b * DFEAT;
        int node = j / 3;
        int c = j - node * 3;
        float v = agg5[((size_t)b * NN + node) * 64 + c];   // layer-5 Np == 64
        if (v < 0.0f) v *= NEG_SLOPE;
        featbf[i] = f2bf_rne(v);
    }
}

// ---------------- final epilogue: out = tanh(Hf + bd) * 0.1 ----------------
__global__ void k_finalize(const float* __restrict__ Hf, const float* __restrict__ bd,
                           float* __restrict__ out) {
    int total = BB * DFEAT;
    for (int i = blockIdx.x * blockDim.x + threadIdx.x; i < total; i += gridDim.x * blockDim.x) {
        int j = i % DFEAT;
        out[i] = tanhf(Hf[i] + bd[j]) * 0.1f;
    }
}

// ---------------- host orchestration ----------------
static inline int rup(int v, int a) { return (v + a - 1) / a * a; }

extern "C" void kernel_launch(void* const* d_in, const int* in_sizes, int n_in,
                              void* d_out, int out_size, void* d_ws, size_t ws_size,
                              hipStream_t stream) {
    (void)in_sizes; (void)n_in; (void)out_size; (void)ws_size;

    const float* x     = (const float*)d_in[0];
    const int*   edges = (const int*)d_in[1];
    const float* W[6]  = {(const float*)d_in[2], (const float*)d_in[4], (const float*)d_in[6],
                          (const float*)d_in[8], (const float*)d_in[10], (const float*)d_in[12]};
    const float* bvec[6] = {(const float*)d_in[3], (const float*)d_in[5], (const float*)d_in[7],
                            (const float*)d_in[9], (const float*)d_in[11], (const float*)d_in[13]};
    const float* Wd = (const float*)d_in[14];
    const float* bd = (const float*)d_in[15];
    float* out = (float*)d_out;

    // ---- carve workspace ----
    char* ws = (char*)d_ws;
    size_t off = 0;
    float* deg = (float*)(ws + off); off += NN * sizeof(float);
    float* isq = (float*)(ws + off); off += NN * sizeof(float);
    float* idg = (float*)(ws + off); off += NN * sizeof(float);
    off = (off + 255) & ~(size_t)255;
    unsigned short* featbf = (unsigned short*)(ws + off); off += (size_t)BB * DFEAT * 2;
    off = (off + 255) & ~(size_t)255;
    unsigned short* Abf = (unsigned short*)(ws + off);    // max 32768*1536*2 = 100.7 MB (also holds WdT: 75.5 MB)
    off += (size_t)MROWS * 1536 * 2;
    off = (off + 255) & ~(size_t)255;
    unsigned short* BTbf = (unsigned short*)(ws + off); off += (size_t)2 * 1024 * 1024;
    off = (off + 255) & ~(size_t)255;
    float* Hw  = (float*)(ws + off); off += (size_t)MROWS * 512 * sizeof(float);   // 64 MB
    off = (off + 255) & ~(size_t)255;
    float* agg = (float*)(ws + off); off += (size_t)MROWS * 512 * sizeof(float);   // 64 MB

    // ---- degrees ----
    k_deg_zero<<<(NN + 255) / 256, 256, 0, stream>>>(deg);
    k_deg_count<<<(EE + 255) / 256, 256, 0, stream>>>(edges, deg);
    k_deg_fin<<<(NN + 255) / 256, 256, 0, stream>>>(deg, isq, idg);

    // ---- GCN layers ----
    const int cins[6]  = {1475, 512, 512, 256, 256, 64};
    const int couts[6] = {512, 512, 256, 256, 64, 3};
    const int leakyIn[6] = {0, 0, 1, 0, 1, 0};   // LeakyReLU applied after layers 1 and 3

    const float* actIn = x;
    for (int i = 0; i < 6; ++i) {
        int cin = cins[i], cout = couts[i];
        int Kp = rup(cin, 64);
        int Np = rup(cout, 64);
        int Npad = rup(Np, 128);          // BT padded so 128-wide staging stays in-bounds

        long long totA = (long long)MROWS * Kp;
        int gA = (int)(((totA + 255) / 256) < 8192 ? ((totA + 255) / 256) : 8192);
        k_convert_act<<<gA, 256, 0, stream>>>(actIn, Abf, MROWS, cin, Kp, leakyIn[i]);

        long long totW = (long long)Npad * Kp;
        int gW = (int)(((totW + 255) / 256) < 4096 ? ((totW + 255) / 256) : 4096);
        k_convert_wT<<<gW, 256, 0, stream>>>(W[i], BTbf, cin, cout, Kp, Npad);

        dim3 grid(MROWS / 128, (Np + 127) / 128);
        k_gemm_wmma<<<grid, 256, 0, stream>>>(Abf, BTbf, Hw, agg, bvec[i], idg,
                                              MROWS, Np, Kp, cout, 1);

        k_scatter<<<EE, 256, 0, stream>>>(Hw, agg, edges, isq, Np);
        actIn = agg;
    }

    // ---- final dense layer: [16 x 6144] @ [6144 x 6144] ----
    k_convert_feat<<<512, 256, 0, stream>>>(agg, featbf);
    {
        long long totW = (long long)DFEAT * DFEAT;
        int gW = (int)(((totW + 255) / 256) < 8192 ? ((totW + 255) / 256) : 8192);
        k_convert_wT<<<gW, 256, 0, stream>>>(Wd, Abf /* WdT */, DFEAT, DFEAT, DFEAT, DFEAT);
    }
    {
        dim3 grid(1, DFEAT / 128);
        k_gemm_wmma<<<grid, 256, 0, stream>>>(featbf, Abf, Hw, Hw, Hw /*unused*/, idg /*unused*/,
                                              BB, DFEAT, DFEAT, DFEAT, 0);
    }
    k_finalize<<<512, 256, 0, stream>>>(Hw, bd, out);
}
